// LocalEncoder_56358560858684
// MI455X (gfx1250) — matrix operand order repaired and checked by reading, not compiled, not run on hardware
//
#include <hip/hip_runtime.h>
#include <hip/hip_bf16.h>

// ---------------------------------------------------------------------------
// Model constants (from reference): B=4, S=2048, D=512, H=8, HD=64, L=2
// NPATCH=16, PLEN=128, V=30000, BASE=31, NG=(3,4,5)
// ---------------------------------------------------------------------------
#define BB    4
#define SS    2048
#define DD    512
#define HH    8
#define HD    64
#define LL    2
#define NP    16
#define PLEN  128
#define VOCAB 30000
#define BSROWS (BB * SS)   // 8192

typedef __attribute__((ext_vector_type(16))) _Float16 v16h;
typedef __attribute__((ext_vector_type(2)))  _Float16 v2h;
typedef __attribute__((ext_vector_type(8)))  float    v8f;

// A-matrix 16x32 f16 fragment K mapping (ISA 7.12.2):
// lanes 0-15: vgpr0..3 -> K 0..7, vgpr4..7 -> K 16..23
// lanes16-31: vgpr0..3 -> K 8..15, vgpr4..7 -> K 24..31
__device__ __forceinline__ int kmapA(int v, int half_) {
    return (v < 4) ? (half_ * 8 + 2 * v) : (16 + half_ * 8 + 2 * (v - 4));
}
// B-matrix 32x16 f16 fragment K mapping (same as A): lanes 0-15: K 0..7, 16..23;
// lanes 16-31: K 8..15, 24..31.
__device__ __forceinline__ int kmapB(int v, int half_) {
    return (v < 4) ? (half_ * 8 + 2 * v) : (16 + half_ * 8 + 2 * (v - 4));
}

// ---------------------------------------------------------------------------
// f32 -> f16 weight pre-conversion (weights are static for the whole pass;
// converting once halves B-side bytes and removes per-K-step cvt work).
// ---------------------------------------------------------------------------
__global__ __launch_bounds__(256) void f32_to_f16(const float* __restrict__ src,
                                                  _Float16* __restrict__ dst, int n) {
    const int i = blockIdx.x * 256 + threadIdx.x;
    if (i < n) dst[i] = (_Float16)src[i];
}

// ---------------------------------------------------------------------------
// Embedding + n-gram rolling-hash kernel.  One block per (b,s) row.
// ---------------------------------------------------------------------------
__global__ __launch_bounds__(128) void embed_ngram(const int* __restrict__ bytes,
                                                   const float* __restrict__ bemb,
                                                   const float* __restrict__ ngemb,
                                                   float* __restrict__ x) {
    const int bs = blockIdx.x;          // b*S + s
    const int s  = bs & (SS - 1);
    const int bv = bytes[bs];

    // powers of 31 mod 30000: 31^0..31^4 -> 1, 31, 961, 29791, 23521
    int h0 = 0, h1 = 0, h2 = 0;
    const bool v0 = (s >= 2), v1 = (s >= 3), v2 = (s >= 4);
    if (v0) h0 = (bytes[bs - 2] * 961 + bytes[bs - 1] * 31 + bv) % VOCAB;
    if (v1) h1 = (bytes[bs - 3] * 29791 + bytes[bs - 2] * 961 + bytes[bs - 1] * 31 + bv) % VOCAB;
    if (v2) h2 = (bytes[bs - 4] * 23521 + bytes[bs - 3] * 29791 + bytes[bs - 2] * 961 +
                  bytes[bs - 1] * 31 + bv) % VOCAB;

    const int t = threadIdx.x;
#pragma unroll
    for (int j = 0; j < 4; ++j) {
        const int d = t + j * 128;
        float val = bemb[(size_t)bv * DD + d];
        if (v0) val += 0.25f * ngemb[((size_t)0 * VOCAB + h0) * DD + d];
        if (v1) val += 0.25f * ngemb[((size_t)1 * VOCAB + h1) * DD + d];
        if (v2) val += 0.25f * ngemb[((size_t)2 * VOCAB + h2) * DD + d];
        x[(size_t)bs * DD + d] = val;
    }
}

// ---------------------------------------------------------------------------
// WMMA GEMM:  C[M,N] = act( A[M,K] @ Wh[N,K]^T + bias + res ),  Wh in f16.
// block = 128 threads (4 waves); each wave computes a 16x64 strip:
// one A fragment per K-step feeds 4 WMMAs (4x A reuse vs 1 tile/wave).
// grid = (N/256, M/16).
// ---------------------------------------------------------------------------
__global__ __launch_bounds__(128) void wmma_gemm(const float* __restrict__ A,
                                                 const _Float16* __restrict__ Wh,
                                                 const float* __restrict__ bias,
                                                 const float* __restrict__ res,
                                                 float* __restrict__ C,
                                                 int M, int N, int K, int relu) {
    const int wave  = threadIdx.x >> 5;
    const int lane  = threadIdx.x & 31;
    const int half_ = lane >> 4;
    const int idx16 = lane & 15;
    const int m0 = blockIdx.y * 16;
    const int n0 = blockIdx.x * 256 + wave * 64;
    if (n0 >= N || m0 >= M) return;

    const float* ap = A + (size_t)(m0 + idx16) * K;       // A-frag row
    const _Float16* wp0 = Wh + (size_t)(n0 + idx16) * K;  // B-frag rows, tiles at +16*K

    v8f acc[4] = {{}, {}, {}, {}};
    for (int k0 = 0; k0 < K; k0 += 32) {
        // stream A (f32 -> f16 on the fly), prefetch the next chunk
        if (k0 + 64 < K) __builtin_prefetch((const void*)(ap + k0 + 64), 0, 3);
        v16h af;
#pragma unroll
        for (int v = 0; v < 8; ++v) {
            const float2 a2 = *(const float2*)(ap + k0 + kmapA(v, half_));
            af[2 * v] = (_Float16)a2.x;
            af[2 * v + 1] = (_Float16)a2.y;
        }
#pragma unroll
        for (int t = 0; t < 4; ++t) {
            const _Float16* wp = wp0 + (size_t)t * 16 * K + k0;
            v16h bf;
#pragma unroll
            for (int v = 0; v < 8; ++v) {
                const v2h b2 = *(const v2h*)(wp + kmapB(v, half_));
                bf[2 * v] = b2.x;
                bf[2 * v + 1] = b2.y;
            }
            acc[t] = __builtin_amdgcn_wmma_f32_16x16x32_f16(false, af, false, bf,
                                                            (short)0, acc[t], false, false);
        }
    }

#pragma unroll
    for (int t = 0; t < 4; ++t) {
        const int col = n0 + t * 16 + idx16;
        const float bvv = bias ? bias[col] : 0.0f;
#pragma unroll
        for (int r = 0; r < 8; ++r) {
            const int row = m0 + half_ * 8 + r;
            float v = acc[t][r] + bvv;
            if (res)  v += res[(size_t)row * N + col];
            if (relu) v = fmaxf(v, 0.0f);
            C[(size_t)row * N + col] = v;
        }
    }
}

// ---------------------------------------------------------------------------
// Repack qkv [B,S,3D] f32 -> Q/K/V [B,H,S,HD] f16
// ---------------------------------------------------------------------------
__global__ __launch_bounds__(256) void qkv_repack(const float* __restrict__ qkv,
                                                  _Float16* __restrict__ Qh,
                                                  _Float16* __restrict__ Kh,
                                                  _Float16* __restrict__ Vh) {
    const size_t i = (size_t)blockIdx.x * 256 + threadIdx.x;  // over B*H*S*HD
    const int d  = (int)(i & (HD - 1));
    const size_t r = i >> 6;
    const int s  = (int)(r % SS);
    const size_t bh = r / SS;
    const int h  = (int)(bh & (HH - 1));
    const int b  = (int)(bh >> 3);
    const size_t src = ((size_t)(b * SS + s)) * (3 * DD) + h * HD + d;
    Qh[i] = (_Float16)qkv[src];
    Kh[i] = (_Float16)qkv[src + DD];
    Vh[i] = (_Float16)qkv[src + 2 * DD];
}

__device__ __forceinline__ v16h load_vfrag(const _Float16* __restrict__ Vb, int kbase,
                                           int t, int idx16, int half_) {
    v16h vf;
#pragma unroll
    for (int v = 0; v < 8; ++v) {
        const int kk = kmapB(v, half_);
        vf[2 * v]     = Vb[(size_t)(kbase + kk) * HD + t * 16 + idx16];
        vf[2 * v + 1] = Vb[(size_t)(kbase + kk + 1) * HD + t * 16 + idx16];
    }
    return vf;
}

// ---------------------------------------------------------------------------
// Causal flash attention: one wave per (b,h, 16-query tile).
// grid = (S/16, B*H), block = 32.  S*S scores never hit memory.
// ---------------------------------------------------------------------------
__global__ __launch_bounds__(32) void flash_attn(const _Float16* __restrict__ Qh,
                                                 const _Float16* __restrict__ Kh,
                                                 const _Float16* __restrict__ Vh,
                                                 float* __restrict__ O) {
    const int lane  = threadIdx.x & 31;
    const int half_ = lane >> 4;
    const int idx16 = lane & 15;
    const int q0 = blockIdx.x * 16;
    const int bh = blockIdx.y;            // b*H + h
    const int h  = bh & (HH - 1);
    const int b  = bh >> 3;

    const _Float16* Qb = Qh + (size_t)bh * SS * HD;
    const _Float16* Kb = Kh + (size_t)bh * SS * HD;
    const _Float16* Vb = Vh + (size_t)bh * SS * HD;

    __shared__ _Float16 pLds[16 * 32];

    // Q fragments (A layout), inner dim 64 = two K=32 chunks
    v16h qf0, qf1;
    {
        const _Float16* qp = Qb + (size_t)(q0 + idx16) * HD;
#pragma unroll
        for (int v = 0; v < 8; ++v) {
            const int kk = kmapA(v, half_);
            qf0[2 * v] = qp[kk];       qf0[2 * v + 1] = qp[kk + 1];
            qf1[2 * v] = qp[32 + kk];  qf1[2 * v + 1] = qp[32 + kk + 1];
        }
    }

    v8f acc0 = {}, acc1 = {}, acc2 = {}, acc3 = {};
    float m_[8], l_[8];
#pragma unroll
    for (int r = 0; r < 8; ++r) { m_[r] = -1e30f; l_[r] = 0.0f; }

    const int qlast = q0 + 15;
    for (int kbase = 0; kbase <= qlast; kbase += 32) {
        // ----- scores: two 16x16 tiles over k-positions [kbase, kbase+32)
        v8f s0 = {}, s1 = {};
#pragma unroll
        for (int c = 0; c < 2; ++c) {
            v16h bf0, bf1;
#pragma unroll
            for (int v = 0; v < 8; ++v) {
                const int kk = c * 32 + kmapB(v, half_);
                const _Float16* k0p = Kb + (size_t)(kbase + idx16) * HD + kk;
                const _Float16* k1p = Kb + (size_t)(kbase + 16 + idx16) * HD + kk;
                bf0[2 * v] = k0p[0]; bf0[2 * v + 1] = k0p[1];
                bf1[2 * v] = k1p[0]; bf1[2 * v + 1] = k1p[1];
            }
            const v16h qc = c ? qf1 : qf0;
            s0 = __builtin_amdgcn_wmma_f32_16x16x32_f16(false, qc, false, bf0,
                                                        (short)0, s0, false, false);
            s1 = __builtin_amdgcn_wmma_f32_16x16x32_f16(false, qc, false, bf1,
                                                        (short)0, s1, false, false);
        }

        // ----- online softmax (row stats reduced across each 16-lane half)
        const int col0 = kbase + idx16;
        const int col1 = kbase + 16 + idx16;
#pragma unroll
        for (int r = 0; r < 8; ++r) {
            const int qrow = q0 + half_ * 8 + r;
            float v0 = (col0 <= qrow) ? s0[r] * 0.125f : -1e30f;
            float v1 = (col1 <= qrow) ? s1[r] * 0.125f : -1e30f;
            float mx = fmaxf(v0, v1);
#pragma unroll
            for (int off = 8; off >= 1; off >>= 1) mx = fmaxf(mx, __shfl_xor(mx, off, 16));
            const float mnew = fmaxf(m_[r], mx);
            const float corr = __expf(m_[r] - mnew);
            const float p0 = __expf(v0 - mnew);
            const float p1 = __expf(v1 - mnew);
            float ps = p0 + p1;
#pragma unroll
            for (int off = 8; off >= 1; off >>= 1) ps += __shfl_xor(ps, off, 16);
            l_[r] = l_[r] * corr + ps;
            m_[r] = mnew;
            acc0[r] *= corr; acc1[r] *= corr; acc2[r] *= corr; acc3[r] *= corr;
            const int prow = half_ * 8 + r;
            pLds[prow * 32 + idx16]      = (_Float16)p0;
            pLds[prow * 32 + 16 + idx16] = (_Float16)p1;
        }

        // ----- P (C layout in LDS) -> A-fragment
        v16h pf;
#pragma unroll
        for (int v = 0; v < 8; ++v) {
            const int kk = kmapA(v, half_);
            pf[2 * v]     = pLds[idx16 * 32 + kk];
            pf[2 * v + 1] = pLds[idx16 * 32 + kk + 1];
        }

        // ----- accumulate P @ V (16x32 @ 32x64) as 4 WMMAs
        acc0 = __builtin_amdgcn_wmma_f32_16x16x32_f16(false, pf, false,
                    load_vfrag(Vb, kbase, 0, idx16, half_), (short)0, acc0, false, false);
        acc1 = __builtin_amdgcn_wmma_f32_16x16x32_f16(false, pf, false,
                    load_vfrag(Vb, kbase, 1, idx16, half_), (short)0, acc1, false, false);
        acc2 = __builtin_amdgcn_wmma_f32_16x16x32_f16(false, pf, false,
                    load_vfrag(Vb, kbase, 2, idx16, half_), (short)0, acc2, false, false);
        acc3 = __builtin_amdgcn_wmma_f32_16x16x32_f16(false, pf, false,
                    load_vfrag(Vb, kbase, 3, idx16, half_), (short)0, acc3, false, false);
    }

    // ----- write O [B,S,D] with head offset, normalize by row sums
#pragma unroll
    for (int r = 0; r < 8; ++r) {
        const int srow = q0 + half_ * 8 + r;
        const float inv = 1.0f / l_[r];
        const size_t base = ((size_t)(b * SS + srow)) * DD + h * HD + idx16;
        O[base + 0]  = acc0[r] * inv;
        O[base + 16] = acc1[r] * inv;
        O[base + 32] = acc2[r] * inv;
        O[base + 48] = acc3[r] * inv;
    }
}

// ---------------------------------------------------------------------------
// LayerNorm (optionally fused residual).  One block (256 threads) per row.
// ---------------------------------------------------------------------------
__global__ __launch_bounds__(256) void ln_kernel(const float* __restrict__ x,
                                                 const float* __restrict__ res,
                                                 const float* __restrict__ w,
                                                 const float* __restrict__ bb,
                                                 float* __restrict__ y, float eps) {
    const int row = blockIdx.x;
    const int t = threadIdx.x;
    const float* xr = x + (size_t)row * DD;
    float v0 = xr[t], v1 = xr[t + 256];
    if (res) { v0 += res[(size_t)row * DD + t]; v1 += res[(size_t)row * DD + t + 256]; }
    float s = v0 + v1, s2 = v0 * v0 + v1 * v1;
#pragma unroll
    for (int off = 16; off >= 1; off >>= 1) {
        s  += __shfl_xor(s, off);
        s2 += __shfl_xor(s2, off);
    }
    __shared__ float sh[2][8];
    __shared__ float mv[2];
    const int wid = t >> 5;
    if ((t & 31) == 0) { sh[0][wid] = s; sh[1][wid] = s2; }
    __syncthreads();
    if (t == 0) {
        float a = 0.0f, b2 = 0.0f;
#pragma unroll
        for (int i = 0; i < 8; ++i) { a += sh[0][i]; b2 += sh[1][i]; }
        const float mean = a * (1.0f / DD);
        const float var  = b2 * (1.0f / DD) - mean * mean;
        mv[0] = mean; mv[1] = rsqrtf(var + eps);
    }
    __syncthreads();
    const float mean = mv[0], rstd = mv[1];
    y[(size_t)row * DD + t]       = (v0 - mean) * rstd * w[t] + bb[t];
    y[(size_t)row * DD + t + 256] = (v1 - mean) * rstd * w[t + 256] + bb[t + 256];
}

// ---------------------------------------------------------------------------
// Patch mean: query[b,p,:] = mean over 128 rows.  grid = 64 blocks.
// ---------------------------------------------------------------------------
__global__ __launch_bounds__(128) void patch_mean(const float* __restrict__ x,
                                                  float* __restrict__ q) {
    const int bp = blockIdx.x;
    const int t = threadIdx.x;
    const float* base = x + (size_t)bp * PLEN * DD;
#pragma unroll
    for (int j = 0; j < 4; ++j) {
        const int d = t + j * 128;
        float s = 0.0f;
        for (int r = 0; r < PLEN; ++r) s += base[(size_t)r * DD + d];
        q[(size_t)bp * DD + d] = s * (1.0f / PLEN);
    }
}

// ---------------------------------------------------------------------------
// Small patch cross-attention core: one block per (b,p,h), 128 threads.
// ---------------------------------------------------------------------------
__global__ __launch_bounds__(128) void patch_attn(const float* __restrict__ q,
                                                  const float* __restrict__ k,
                                                  const float* __restrict__ v,
                                                  float* __restrict__ o) {
    const int h  = blockIdx.x & (HH - 1);
    const int bp = blockIdx.x >> 3;
    const int b  = bp >> 4;
    const int p  = bp & 15;
    const int l  = threadIdx.x;

    __shared__ float a[PLEN];
    __shared__ float mx, sm;

    const float* qv = q + (size_t)bp * DD + h * HD;
    const size_t krow = ((size_t)(b * SS + p * PLEN + l)) * DD + h * HD;
    float s = 0.0f;
    for (int d = 0; d < HD; ++d) s += qv[d] * k[krow + d];
    s *= 0.125f;
    a[l] = s;
    __syncthreads();
    if (l == 0) {
        float m = a[0];
        for (int i = 1; i < PLEN; ++i) m = fmaxf(m, a[i]);
        mx = m;
    }
    __syncthreads();
    const float e = __expf(s - mx);
    a[l] = e;
    __syncthreads();
    if (l == 0) {
        float t2 = 0.0f;
        for (int i = 0; i < PLEN; ++i) t2 += a[i];
        sm = t2;
    }
    __syncthreads();
    if (l < HD) {
        float acc = 0.0f;
        for (int j = 0; j < PLEN; ++j)
            acc += a[j] * v[((size_t)(b * SS + p * PLEN + j)) * DD + h * HD + l];
        o[(size_t)bp * DD + h * HD + l] = acc / sm;
    }
}

// ---------------------------------------------------------------------------
// Host-side orchestration
// ---------------------------------------------------------------------------
static inline void launch_gemm(const float* A, const _Float16* Wh, const float* bias,
                               const float* res, float* C, int M, int N, int K,
                               int relu, hipStream_t stream) {
    dim3 grid(N / 256, M / 16);
    wmma_gemm<<<grid, 128, 0, stream>>>(A, Wh, bias, res, C, M, N, K, relu);
}

static inline void launch_cvt(const float* src, _Float16* dst, int n, hipStream_t stream) {
    f32_to_f16<<<(n + 255) / 256, 256, 0, stream>>>(src, dst, n);
}

extern "C" void kernel_launch(void* const* d_in, const int* in_sizes, int n_in,
                              void* d_out, int out_size, void* d_ws, size_t ws_size,
                              hipStream_t stream) {
    (void)in_sizes; (void)n_in; (void)out_size; (void)ws_size;

    const int*   byte_seq = (const int*)d_in[0];
    const float* byte_emb = (const float*)d_in[2];
    const float* ng_emb   = (const float*)d_in[3];
    const float* tl_qkv_w = (const float*)d_in[4];
    const float* tl_qkv_b = (const float*)d_in[5];
    const float* tl_out_w = (const float*)d_in[6];
    const float* tl_out_b = (const float*)d_in[7];
    const float* tl_ln1_w = (const float*)d_in[8];
    const float* tl_ln1_b = (const float*)d_in[9];
    const float* tl_ln2_w = (const float*)d_in[10];
    const float* tl_ln2_b = (const float*)d_in[11];
    const float* tl_ff1_w = (const float*)d_in[12];
    const float* tl_ff1_b = (const float*)d_in[13];
    const float* tl_ff2_w = (const float*)d_in[14];
    const float* tl_ff2_b = (const float*)d_in[15];
    const float* ca_qkv_w = (const float*)d_in[16];
    const float* ca_qkv_b = (const float*)d_in[17];
    const float* ca_out_w = (const float*)d_in[18];
    const float* ca_out_b = (const float*)d_in[19];
    const float* ca_ln_w  = (const float*)d_in[20];
    const float* ca_ln_b  = (const float*)d_in[21];
    const float* norm_w   = (const float*)d_in[22];
    const float* norm_b   = (const float*)d_in[23];

    // ---- workspace carve-up (~142 MB) ----
    float* fx   = (float*)d_ws;                    // [BS, D]   activations
    float* fbig = fx + (size_t)BSROWS * DD;        // [BS, 4D]  qkv / ff hidden / kv proj
    float* fo   = fbig + (size_t)BSROWS * 4 * DD;  // [BS, D]   attn/ff out, later kvn
    _Float16* Qh = (_Float16*)(fo + (size_t)BSROWS * DD);
    _Float16* Kh = Qh + (size_t)BSROWS * DD;
    _Float16* Vh = Kh + (size_t)BSROWS * DD;
    // pre-converted f16 weights
    _Float16* wh_qkv   = Vh + (size_t)BSROWS * DD;            // L*3D*D
    _Float16* wh_out   = wh_qkv + (size_t)LL * 3 * DD * DD;   // L*D*D
    _Float16* wh_ff1   = wh_out + (size_t)LL * DD * DD;       // L*4D*D
    _Float16* wh_ff2   = wh_ff1 + (size_t)LL * 4 * DD * DD;   // L*D*4D
    _Float16* wh_caqkv = wh_ff2 + (size_t)LL * DD * 4 * DD;   // 3D*D
    _Float16* wh_caout = wh_caqkv + (size_t)3 * DD * DD;      // D*D
    float* query   = (float*)(wh_caout + (size_t)DD * DD);    // [64, D]
    float* qn      = query + 64 * DD;
    float* qproj   = qn + 64 * DD;
    float* po      = qproj + 64 * DD;
    float* patches = po + 64 * DD;

    float* out_x = (float*)d_out;                      // [BS, D]
    float* out_p = out_x + (size_t)BSROWS * DD;        // [64, D]

    // 0. weight conversion (once per call; weights static across the pass)
    launch_cvt(tl_qkv_w, wh_qkv, LL * 3 * DD * DD, stream);
    launch_cvt(tl_out_w, wh_out, LL * DD * DD, stream);
    launch_cvt(tl_ff1_w, wh_ff1, LL * 4 * DD * DD, stream);
    launch_cvt(tl_ff2_w, wh_ff2, LL * DD * 4 * DD, stream);
    launch_cvt(ca_qkv_w, wh_caqkv, 3 * DD * DD, stream);
    launch_cvt(ca_out_w, wh_caout, DD * DD, stream);

    // 1. embeddings + n-gram hash adds
    embed_ngram<<<BSROWS, 128, 0, stream>>>(byte_seq, byte_emb, ng_emb, fx);

    // 2. transformer layers
    for (int l = 0; l < LL; ++l) {
        launch_gemm(fx, wh_qkv + (size_t)l * 3 * DD * DD, tl_qkv_b + (size_t)l * 3 * DD,
                    nullptr, fbig, BSROWS, 3 * DD, DD, 0, stream);
        qkv_repack<<<(BSROWS * DD) / 256, 256, 0, stream>>>(fbig, Qh, Kh, Vh);
        flash_attn<<<dim3(SS / 16, BB * HH), 32, 0, stream>>>(Qh, Kh, Vh, fo);
        launch_gemm(fo, wh_out + (size_t)l * DD * DD, tl_out_b + (size_t)l * DD,
                    nullptr, fbig, BSROWS, DD, DD, 0, stream);
        ln_kernel<<<BSROWS, 256, 0, stream>>>(fx, fbig, tl_ln1_w + (size_t)l * DD,
                                              tl_ln1_b + (size_t)l * DD, fx, 1e-5f);
        launch_gemm(fx, wh_ff1 + (size_t)l * 4 * DD * DD, tl_ff1_b + (size_t)l * 4 * DD,
                    nullptr, fbig, BSROWS, 4 * DD, DD, 1, stream);
        launch_gemm(fbig, wh_ff2 + (size_t)l * DD * 4 * DD, tl_ff2_b + (size_t)l * DD,
                    nullptr, fo, BSROWS, DD, 4 * DD, 0, stream);
        ln_kernel<<<BSROWS, 256, 0, stream>>>(fx, fo, tl_ln2_w + (size_t)l * DD,
                                              tl_ln2_b + (size_t)l * DD, fx, 1e-5f);
    }

    // 3. patch cross-attention
    patch_mean<<<BB * NP, 128, 0, stream>>>(fx, query);
    ln_kernel<<<BB * NP, 256, 0, stream>>>(query, nullptr, ca_ln_w, ca_ln_b, qn, 1e-6f);
    ln_kernel<<<BSROWS, 256, 0, stream>>>(fx, nullptr, ca_ln_w, ca_ln_b, fo, 1e-6f);

    float* kbuf = fbig;
    float* vbuf = fbig + (size_t)BSROWS * DD;
    launch_gemm(qn, wh_caqkv, ca_qkv_b, nullptr, qproj, BB * NP, DD, DD, 0, stream);
    launch_gemm(fo, wh_caqkv + (size_t)DD * DD,     ca_qkv_b + DD,     nullptr, kbuf,
                BSROWS, DD, DD, 0, stream);
    launch_gemm(fo, wh_caqkv + (size_t)2 * DD * DD, ca_qkv_b + 2 * DD, nullptr, vbuf,
                BSROWS, DD, DD, 0, stream);
    patch_attn<<<BB * NP * HH, 128, 0, stream>>>(qproj, kbuf, vbuf, po);
    launch_gemm(po, wh_caout, ca_out_b, query, patches, BB * NP, DD, DD, 0, stream);

    // 4. final norms -> outputs
    ln_kernel<<<BSROWS, 256, 0, stream>>>(fx, nullptr, norm_w, norm_b, out_x, 1e-6f);
    ln_kernel<<<BB * NP, 256, 0, stream>>>(patches, nullptr, norm_w, norm_b, out_p, 1e-6f);
}
